// MultiHeadAttention_49804440764576
// MI455X (gfx1250) — matrix operand order repaired and checked by reading, not compiled, not run on hardware
//
#include <hip/hip_runtime.h>
#include <hip/hip_bf16.h>

typedef __attribute__((ext_vector_type(16))) _Float16 v16h;
typedef __attribute__((ext_vector_type(8)))  _Float16 v8h;
typedef __attribute__((ext_vector_type(4)))  _Float16 v4h;
typedef __attribute__((ext_vector_type(8)))  float    v8f;

#define BATCH 2
#define SEQ   2048
#define DIM   1024
#define NH    16
#define HS    64

// LDS stage geometry for the GEMMs (halves): A 64x40, B 256x40 (pitch 40 kills
// the 16-lane bank collision of a 32-half pitch while keeping 16B alignment).
#define APITCH 40
#define A_HALVES (64 * APITCH)             // 2560
#define STAGE_HALVES (A_HALVES + 256 * APITCH)  // 12800 halves = 25600 B

__device__ __forceinline__ v16h combine16(v8h lo, v8h hi) {
  v16h r;
#pragma unroll
  for (int i = 0; i < 8; ++i) { r[i] = lo[i]; r[i + 8] = hi[i]; }
  return r;
}

__device__ __forceinline__ v8f wmma_f16(v16h a, v16h b, v8f c) {
  // emits v_wmma_f32_16x16x32_f16
  return __builtin_amdgcn_wmma_f32_16x16x32_f16(false, a, false, b, (short)0, c,
                                                false, false);
}

// CDNA5 async copy: global -> LDS, 16 bytes per lane, tracked by ASYNCcnt.
__device__ __forceinline__ void async_copy16(unsigned lds_off, const void* gaddr) {
  asm volatile("global_load_async_to_lds_b128 %0, %1, off"
               :: "v"(lds_off), "v"(gaddr) : "memory");
}
__device__ __forceinline__ void wait_async_le5() {
  asm volatile("s_wait_asynccnt 0x5" ::: "memory");
}
__device__ __forceinline__ void wait_async_0() {
  asm volatile("s_wait_asynccnt 0x0" ::: "memory");
}

// ---------------- conversion / layout kernels ----------------

__global__ __launch_bounds__(256) void k_cvt_f16(const float* __restrict__ in,
                                                 _Float16* __restrict__ out,
                                                 int n4) {
  int i = blockIdx.x * 256 + threadIdx.x;
  if (i < n4) {
    float4 v = ((const float4*)in)[i];
    v4h h;
    h[0] = (_Float16)v.x; h[1] = (_Float16)v.y;
    h[2] = (_Float16)v.z; h[3] = (_Float16)v.w;
    ((v4h*)out)[i] = h;
  }
}

// in: rows x cols (f32, row-major) -> out: cols x rows (f16, row-major)
__global__ __launch_bounds__(256) void k_transpose_f16(const float* __restrict__ in,
                                                       _Float16* __restrict__ out,
                                                       int rows, int cols) {
  int i = blockIdx.x * 256 + threadIdx.x;
  if (i < rows * cols) {
    int r = i / cols;
    int c = i - r * cols;
    out[(size_t)c * rows + r] = (_Float16)in[i];
  }
}

// ---------------- shared GEMM core: async double-buffered LDS pipeline -------
// Workgroup tile 64(M) x 256(N); 8 waves as 2x4, each wave 32x64 (8 WMMA/K-step).

__device__ __forceinline__ void gemm_stage_issue(const _Float16* __restrict__ A,
                                                 const _Float16* __restrict__ Bt,
                                                 int m0blk, int n0blk, int k,
                                                 unsigned lbase, int tid) {
  // A tile: 64 rows x 32 halves = 256 x 16B chunks, one per thread
  {
    const int row = tid >> 2, co = (tid & 3) * 8;
    async_copy16(lbase + (unsigned)(row * APITCH + co) * 2,
                 A + (size_t)(m0blk + row) * DIM + k + co);
  }
  // B tile: 256 rows x 32 halves = 1024 chunks, four per thread
#pragma unroll
  for (int i = 0; i < 4; ++i) {
    const int c = tid + i * 256;
    const int row = c >> 2, co = (c & 3) * 8;
    async_copy16(lbase + (unsigned)(A_HALVES + row * APITCH + co) * 2,
                 Bt + (size_t)(n0blk + row) * DIM + k + co);
  }
}

__device__ __forceinline__ void gemm_core(const _Float16* __restrict__ A,
                                          const _Float16* __restrict__ Bt,
                                          _Float16* sh,          // [2*STAGE_HALVES]
                                          int m0blk, int n0blk,
                                          v8f acc[2][4]) {
  const int tid = threadIdx.x;
  const int lane = tid & 31;
  const int wave = tid >> 5;
  const int r = lane & 15, hi = lane >> 4;
  const int wm = (wave >> 2) * 32;      // wave M offset in block tile
  const int wn = (wave & 3) * 64;       // wave N offset in block tile
  const unsigned lbase = (unsigned)(uintptr_t)sh;

  gemm_stage_issue(A, Bt, m0blk, n0blk, 0, lbase, tid);

  for (int kk = 0; kk < DIM / 32; ++kk) {
    if (kk < DIM / 32 - 1) {
      gemm_stage_issue(A, Bt, m0blk, n0blk, (kk + 1) * 32,
                       lbase + (unsigned)(((kk + 1) & 1) * STAGE_HALVES * 2), tid);
      wait_async_le5();                 // oldest 5 (current stage) have landed
    } else {
      wait_async_0();
    }
    __syncthreads();                    // stage visible to all waves

    const _Float16* buf = sh + (kk & 1) * STAGE_HALVES;
    v16h af[2], bf[4];
#pragma unroll
    for (int mt = 0; mt < 2; ++mt) {
      const _Float16* p = buf + (wm + mt * 16 + r) * APITCH + hi * 8;
      af[mt] = combine16(*(const v8h*)p, *(const v8h*)(p + 16));
    }
#pragma unroll
    for (int nt = 0; nt < 4; ++nt) {
      const _Float16* p = buf + A_HALVES + (wn + nt * 16 + r) * APITCH + hi * 8;
      bf[nt] = combine16(*(const v8h*)p, *(const v8h*)(p + 16));
    }
#pragma unroll
    for (int mt = 0; mt < 2; ++mt)
#pragma unroll
      for (int nt = 0; nt < 4; ++nt)
        acc[mt][nt] = wmma_f16(af[mt], bf[nt], acc[mt][nt]);

    __syncthreads();                    // reads done before buffer is rewritten
  }
}

// ---------------- GEMM 1: X(4096x1024) * Wqkv -> scatter Q/K/Vt ----------------

__global__ __launch_bounds__(256) void k_gemm_qkv(const _Float16* __restrict__ A,
                                                  const _Float16* __restrict__ Bt,
                                                  const float* __restrict__ bias,
                                                  _Float16* __restrict__ Qb,
                                                  _Float16* __restrict__ Kb,
                                                  _Float16* __restrict__ Vt) {
  __shared__ __align__(16) _Float16 sh[2 * STAGE_HALVES];
  const int lane = threadIdx.x & 31;
  const int wave = threadIdx.x >> 5;
  const int r = lane & 15, hi = lane >> 4;
  const int m0 = blockIdx.x * 64 + (wave >> 2) * 32;
  const int n0 = blockIdx.y * 256 + (wave & 3) * 64;

  v8f acc[2][4] = {};
  gemm_core(A, Bt, sh, blockIdx.x * 64, blockIdx.y * 256, acc);

#pragma unroll
  for (int mt = 0; mt < 2; ++mt) {
#pragma unroll
    for (int nt = 0; nt < 4; ++nt) {
      const int gn = n0 + nt * 16 + r;
      const float bia = bias[gn];
#pragma unroll
      for (int i = 0; i < 8; ++i) {
        const int gm = m0 + mt * 16 + i + 8 * hi;
        const int bb = gm >> 11;            // / SEQ
        const int tt = gm & (SEQ - 1);
        const float v = acc[mt][nt][i] + bia;
        if (gn < DIM) {                     // Q (scaled by 1/sqrt(HS))
          const int h = gn >> 6, hs = gn & 63;
          Qb[(((size_t)bb * NH + h) * SEQ + tt) * HS + hs] = (_Float16)(v * 0.125f);
        } else if (gn < 2 * DIM) {          // K
          const int g = gn - DIM, h = g >> 6, hs = g & 63;
          Kb[(((size_t)bb * NH + h) * SEQ + tt) * HS + hs] = (_Float16)v;
        } else {                            // V stored transposed (B,H,HS,T)
          const int g = gn - 2 * DIM, h = g >> 6, hs = g & 63;
          Vt[(((size_t)bb * NH + h) * HS + hs) * SEQ + tt] = (_Float16)v;
        }
      }
    }
  }
}

// ---------------- Flash attention: one wave per 16-query tile ----------------

__global__ __launch_bounds__(256) void k_attn(const _Float16* __restrict__ Qb,
                                              const _Float16* __restrict__ Kb,
                                              const _Float16* __restrict__ Vt,
                                              _Float16* __restrict__ O16) {
  __shared__ __align__(16) _Float16 plds[8][512];   // per-wave 16x32 P tile
  const int lane = threadIdx.x & 31;
  const int wave = threadIdx.x >> 5;
  const int r = lane & 15, hi = lane >> 4;
  const int bh = blockIdx.x >> 4;                   // 16 query-blocks per (b,h)
  const int blk = blockIdx.x & 15;
  const int b = bh >> 4, h = bh & 15;
  const int q0 = blk * 128 + wave * 16;

  const _Float16* Qbh = Qb + (size_t)bh * SEQ * HS;
  const _Float16* Kbh = Kb + (size_t)bh * SEQ * HS;
  const _Float16* Vbh = Vt + (size_t)bh * HS * SEQ;

  v16h qf[2];
#pragma unroll
  for (int c = 0; c < 2; ++c) {
    const _Float16* p = Qbh + (size_t)(q0 + r) * HS + c * 32 + hi * 8;
    qf[c] = combine16(*(const v8h*)p, *(const v8h*)(p + 16));
  }

  v8f o[4] = {};
  float m[8], l[8];
#pragma unroll
  for (int i = 0; i < 8; ++i) { m[i] = -1e30f; l[i] = 0.0f; }

  const int nk = q0 + 16;                           // causal: keys [0, q0+16)
  for (int k0 = 0; k0 < nk; k0 += 32) {
    v8f s0 = {}, s1 = {};
#pragma unroll
    for (int c = 0; c < 2; ++c) {
      const _Float16* p0 = Kbh + (size_t)(k0 + r) * HS + c * 32 + hi * 8;
      v16h kf0 = combine16(*(const v8h*)p0, *(const v8h*)(p0 + 16));
      s0 = wmma_f16(qf[c], kf0, s0);
      const _Float16* p1 = Kbh + (size_t)(k0 + 16 + r) * HS + c * 32 + hi * 8;
      v16h kf1 = combine16(*(const v8h*)p1, *(const v8h*)(p1 + 16));
      s1 = wmma_f16(qf[c], kf1, s1);
    }

    if (k0 + 31 > q0) {                             // causal mask (partial block)
      const int c0 = k0 + r;
#pragma unroll
      for (int i = 0; i < 8; ++i) {
        const int row = q0 + i + 8 * hi;
        if (c0 > row)      s0[i] = -1e30f;
        if (c0 + 16 > row) s1[i] = -1e30f;
      }
    }

    float sc[8];
#pragma unroll
    for (int i = 0; i < 8; ++i) {
      float t = fmaxf(s0[i], s1[i]);
      t = fmaxf(t, __shfl_xor(t, 8, 32));
      t = fmaxf(t, __shfl_xor(t, 4, 32));
      t = fmaxf(t, __shfl_xor(t, 2, 32));
      t = fmaxf(t, __shfl_xor(t, 1, 32));
      const float mn = fmaxf(m[i], t);
      sc[i] = __expf(m[i] - mn);
      m[i] = mn;
      const float p0 = __expf(s0[i] - mn);
      const float p1 = __expf(s1[i] - mn);
      plds[wave][(i + 8 * hi) * 32 + r]      = (_Float16)p0;
      plds[wave][(i + 8 * hi) * 32 + r + 16] = (_Float16)p1;
      float s = p0 + p1;
      s += __shfl_xor(s, 8, 32);
      s += __shfl_xor(s, 4, 32);
      s += __shfl_xor(s, 2, 32);
      s += __shfl_xor(s, 1, 32);
      l[i] = l[i] * sc[i] + s;
    }
#pragma unroll
    for (int nt = 0; nt < 4; ++nt)
#pragma unroll
      for (int i = 0; i < 8; ++i) o[nt][i] *= sc[i];

    __builtin_amdgcn_wave_barrier();                // LDS stores before loads
    const _Float16* pp = &plds[wave][r * 32 + hi * 8];
    const v16h pf = combine16(*(const v8h*)pp, *(const v8h*)(pp + 16));
    __builtin_amdgcn_wave_barrier();                // loads before next stores

#pragma unroll
    for (int nt = 0; nt < 4; ++nt) {
      const _Float16* vp = Vbh + (size_t)(nt * 16 + r) * SEQ + k0 + hi * 8;
      v16h vf = combine16(*(const v8h*)vp, *(const v8h*)(vp + 16));
      o[nt] = wmma_f16(pf, vf, o[nt]);
    }
  }

  _Float16* obase = O16 + ((size_t)(b * SEQ + q0)) * DIM + h * HS;
#pragma unroll
  for (int i = 0; i < 8; ++i) {
    const float inv = 1.0f / l[i];
    _Float16* po = obase + (size_t)(i + 8 * hi) * DIM;
#pragma unroll
    for (int nt = 0; nt < 4; ++nt)
      po[nt * 16 + r] = (_Float16)(o[nt][i] * inv);
  }
}

// ---------------- GEMM 2: O(4096x1024) * Wout + bias -> f32 out ----------------

__global__ __launch_bounds__(256) void k_gemm_out(const _Float16* __restrict__ A,
                                                  const _Float16* __restrict__ Bt,
                                                  const float* __restrict__ bias,
                                                  float* __restrict__ out) {
  __shared__ __align__(16) _Float16 sh[2 * STAGE_HALVES];
  const int lane = threadIdx.x & 31;
  const int wave = threadIdx.x >> 5;
  const int r = lane & 15, hi = lane >> 4;
  const int m0 = blockIdx.x * 64 + (wave >> 2) * 32;
  const int n0 = blockIdx.y * 256 + (wave & 3) * 64;

  v8f acc[2][4] = {};
  gemm_core(A, Bt, sh, blockIdx.x * 64, blockIdx.y * 256, acc);

#pragma unroll
  for (int mt = 0; mt < 2; ++mt) {
#pragma unroll
    for (int nt = 0; nt < 4; ++nt) {
      const int gn = n0 + nt * 16 + r;
      const float bia = bias[gn];
#pragma unroll
      for (int i = 0; i < 8; ++i) {
        const int gm = m0 + mt * 16 + i + 8 * hi;
        out[(size_t)gm * DIM + gn] = acc[mt][nt][i] + bia;
      }
    }
  }
}

// ---------------- host launcher ----------------

extern "C" void kernel_launch(void* const* d_in, const int* in_sizes, int n_in,
                              void* d_out, int out_size, void* d_ws, size_t ws_size,
                              hipStream_t stream) {
  const float* x     = (const float*)d_in[0];
  const float* w_qkv = (const float*)d_in[1];
  const float* b_qkv = (const float*)d_in[2];
  const float* w_out = (const float*)d_in[3];
  const float* b_out = (const float*)d_in[4];
  float* out = (float*)d_out;

  char* ws = (char*)d_ws;
  _Float16* x16   = (_Float16*)(ws + 0);          //  8 MB  (4096x1024)
  _Float16* wqkvT = (_Float16*)(ws + 8388608);    //  6 MB  (3072x1024)
  _Float16* woutT = (_Float16*)(ws + 14680064);   //  2 MB  (1024x1024)
  _Float16* Qb    = (_Float16*)(ws + 16777216);   //  8 MB  (B,H,T,HS)
  _Float16* Kb    = (_Float16*)(ws + 25165824);   //  8 MB  (B,H,T,HS)
  _Float16* Vt    = (_Float16*)(ws + 33554432);   //  8 MB  (B,H,HS,T)
  _Float16* O16   = (_Float16*)(ws + 41943040);   //  8 MB  (4096x1024)

  const int nx4 = BATCH * SEQ * DIM / 4;
  k_cvt_f16<<<(nx4 + 255) / 256, 256, 0, stream>>>(x, x16, nx4);
  k_transpose_f16<<<(DIM * 3 * DIM + 255) / 256, 256, 0, stream>>>(w_qkv, wqkvT, DIM, 3 * DIM);
  k_transpose_f16<<<(DIM * DIM + 255) / 256, 256, 0, stream>>>(w_out, woutT, DIM, DIM);

  k_gemm_qkv<<<dim3(64, 12), 256, 0, stream>>>(x16, wqkvT, b_qkv, Qb, Kb, Vt);
  k_attn<<<dim3(BATCH * NH * (SEQ / 128)), 256, 0, stream>>>(Qb, Kb, Vt, O16);
  k_gemm_out<<<dim3(64, 4), 256, 0, stream>>>(O16, woutT, b_out, out);
}
